// Model_14216341749907
// MI455X (gfx1250) — matrix operand order repaired and checked by reading, not compiled, not run on hardware
//
#include <hip/hip_runtime.h>

// ---------------------------------------------------------------------------
// CDNA5 (gfx1250) implementation of the CO2-Net style reference.
// Heavy conv1d(k=3) layers -> f16 WMMA GEMMs (v_wmma_f32_16x16x32_f16) with
// gfx1250 async global->LDS copies (ASYNCcnt) for the tile pipeline.
// ---------------------------------------------------------------------------

typedef __attribute__((ext_vector_type(16))) _Float16 v16h;
typedef __attribute__((ext_vector_type(8)))  _Float16 v8h;
typedef __attribute__((ext_vector_type(2)))  _Float16 v2h;
typedef __attribute__((ext_vector_type(8)))  float    v8f;
typedef __attribute__((ext_vector_type(4)))  int      v4i;

#define FEAT  1024
#define TLEN  1024
#define NBAT  16
#define NCLS  20
#define KTOT  3072   // 3 * FEAT

#define GLOBAL_AS __attribute__((address_space(1)))
#define LDS_AS    __attribute__((address_space(3)))

#if __has_builtin(__builtin_amdgcn_global_load_async_to_lds_b128) && \
    __has_builtin(__builtin_amdgcn_s_wait_asynccnt)
#define USE_ASYNC_LDS 1
#else
#define USE_ASYNC_LDS 0
#endif

// ---------------------------------------------------------------------------
// Pack x (n,t,2f) f32 -> rgb16/flow16 as [n][t][f] f16
// ---------------------------------------------------------------------------
__global__ __launch_bounds__(256) void pack_x_kernel(
    const float* __restrict__ x, _Float16* __restrict__ rgb, _Float16* __restrict__ flow)
{
    size_t id = (size_t)blockIdx.x * 256 + threadIdx.x;       // over NBAT*TLEN*(FEAT/4)
    size_t total = (size_t)NBAT * TLEN * (FEAT / 4);
    if (id >= total) return;
    size_t f4 = id % (FEAT / 4);
    size_t nt = id / (FEAT / 4);
    const float* px = x + nt * (2 * FEAT) + f4 * 4;
    float4 r  = *(const float4*)px;
    float4 fl = *(const float4*)(px + FEAT);
    union { _Float16 h[4]; uint2 u; } ur, uf;
    ur.h[0] = (_Float16)r.x;  ur.h[1] = (_Float16)r.y;
    ur.h[2] = (_Float16)r.z;  ur.h[3] = (_Float16)r.w;
    uf.h[0] = (_Float16)fl.x; uf.h[1] = (_Float16)fl.y;
    uf.h[2] = (_Float16)fl.z; uf.h[3] = (_Float16)fl.w;
    *(uint2*)(rgb  + nt * FEAT + f4 * 4) = ur.u;
    *(uint2*)(flow + nt * FEAT + f4 * 4) = uf.u;
}

// ---------------------------------------------------------------------------
// Pack conv weight (fo, fi, 3) f32 -> [fo][k*1024+fi] f16 (GEMM-K contiguous)
// ---------------------------------------------------------------------------
__global__ __launch_bounds__(256) void pack_w_kernel(
    const float* __restrict__ w, _Float16* __restrict__ wp)
{
    int id = blockIdx.x * 256 + threadIdx.x;                  // over FEAT*KTOT
    if (id >= FEAT * KTOT) return;
    int kk = id % KTOT;
    int fo = id / KTOT;
    int kb = kk >> 10;
    int fi = kk & 1023;
    wp[(size_t)fo * KTOT + kk] = (_Float16)w[(size_t)fo * KTOT + fi * 3 + kb];
}

// ---------------------------------------------------------------------------
// Temporal mean of x over t: means[n][c] for c in [0,2048)
// ---------------------------------------------------------------------------
__global__ __launch_bounds__(256) void mean_kernel(
    const float* __restrict__ x, float* __restrict__ means)
{
    int n = blockIdx.x, tid = threadIdx.x;
    float acc[8];
#pragma unroll
    for (int q = 0; q < 8; ++q) acc[q] = 0.f;
    for (int t = 0; t < TLEN; ++t) {
        const float* row = x + ((size_t)n * TLEN + t) * (2 * FEAT);
#pragma unroll
        for (int q = 0; q < 8; ++q) acc[q] += row[tid + 256 * q];
    }
#pragma unroll
    for (int q = 0; q < 8; ++q)
        means[(size_t)n * 2048 + tid + 256 * q] = acc[q] * (1.f / 1024.f);
}

// ---------------------------------------------------------------------------
// g = relu(center-tap GEMV on means + bias) for both streams
// ---------------------------------------------------------------------------
__global__ __launch_bounds__(256) void gvec_kernel(
    const float* __restrict__ gw_r, const float* __restrict__ gb_r,
    const float* __restrict__ gw_f, const float* __restrict__ gb_f,
    const float* __restrict__ means,
    float* __restrict__ g_rgb, float* __restrict__ g_flow)
{
    int id = blockIdx.x * 256 + threadIdx.x;                  // NBAT*FEAT
    int n = id >> 10, fo = id & 1023;
    const float* mr = means + (size_t)n * 2048;
    const float* mf = mr + FEAT;
    const float* wr = gw_r + (size_t)fo * KTOT + 1;
    const float* wf = gw_f + (size_t)fo * KTOT + 1;
    float ar = 0.f, af = 0.f;
    for (int fi = 0; fi < FEAT; ++fi) {
        ar += wr[fi * 3] * mr[fi];
        af += wf[fi * 3] * mf[fi];
    }
    ar += gb_r[fo]; af += gb_f[fo];
    g_rgb[id]  = ar > 0.f ? ar : 0.f;
    g_flow[id] = af > 0.f ? af : 0.f;
}

// ---------------------------------------------------------------------------
// Big kernel: Y[n][t][fo] = conv1d_k3(X, W) with WMMA.
//   mode 0: out = relu(Y + bias)                       (encoder stage-1)
//   mode 1: out = sigmoid(g[n][fo]*relu(Y+bias)) * Gf  (CCA)
// Tile: 128(fo) x 128(t) per workgroup, K in steps of 32.
// Pipeline: double-buffered LDS fed by global_load_async_to_lds_b128.
// ---------------------------------------------------------------------------
__global__ __launch_bounds__(256) void conv3_wmma_kernel(
    const _Float16* __restrict__ xp,    // [n][t][f]
    const _Float16* __restrict__ wp,    // [fo][k*1024+fi]
    const float*    __restrict__ bias,  // [f]
    const float*    __restrict__ gv,    // [n][f] or null
    const _Float16* __restrict__ gf,    // [n][t][f] or null
    _Float16*       __restrict__ out,   // [n][t][f]
    int mode)
{
    __shared__ _Float16 smem[20480];    // A0|A1|B0|B1 (5120 f16 each, 80B rows)

    const int tid  = threadIdx.x;
    const int lane = tid & 31;
    const int wave = tid >> 5;
    const int wt   = wave & 3;          // t sub-block (32 wide)
    const int wf   = wave >> 2;         // fo sub-block (64 tall)
    const int tBase  = blockIdx.x * 128;
    const int foBase = blockIdx.y * 128;
    const int n      = blockIdx.z;
    const int m  = lane & 15;
    const int hi = lane >> 4;
    const int ar = tid >> 2;            // LDS staging row (0..63), +64 for u=1
    const int ac = tid & 3;             // 16B chunk within row

    v8f acc[4][2];
#pragma unroll
    for (int i = 0; i < 4; ++i)
#pragma unroll
        for (int j = 0; j < 2; ++j)
#pragma unroll
            for (int e = 0; e < 8; ++e) acc[i][j][e] = 0.f;

#if USE_ASYNC_LDS
    // gfx1250 async global->LDS copy: no VGPR staging, tracked with ASYNCcnt.
    auto issue_tile = [&](int ks, int buf) {
        const int K0  = ks * 32;
        const int kb  = K0 >> 10;
        const int fi0 = K0 & 1023;
        _Float16* A = smem + buf * 5120;
        _Float16* B = smem + 10240 + buf * 5120;
#pragma unroll
        for (int u = 0; u < 2; ++u) {
            int r = ar + 64 * u;
            const _Float16* ga = wp + (size_t)(foBase + r) * KTOT + K0 + ac * 8;
            __builtin_amdgcn_global_load_async_to_lds_b128(
                (GLOBAL_AS v4i*)ga,
                (LDS_AS v4i*)(A + r * 40 + ac * 8), 0, 0);
            int gt = tBase + r + kb - 1;
            _Float16* lb = B + r * 40 + ac * 8;
            if ((unsigned)gt < (unsigned)TLEN) {
                const _Float16* gb = xp + ((size_t)n * TLEN + gt) * FEAT + fi0 + ac * 8;
                __builtin_amdgcn_global_load_async_to_lds_b128(
                    (GLOBAL_AS v4i*)gb,
                    (LDS_AS v4i*)lb, 0, 0);
            } else {
                *(int4*)lb = make_int4(0, 0, 0, 0);   // zero-pad row via ds_store
            }
        }
    };
#else
    // Fallback: register-staged copy.
    int4 areg[2], breg[2];
    auto gload = [&](int ks) {
        const int K0  = ks * 32;
        const int kb  = K0 >> 10;
        const int fi0 = K0 & 1023;
#pragma unroll
        for (int u = 0; u < 2; ++u) {
            int r = ar + 64 * u;
            areg[u] = *(const int4*)(wp + (size_t)(foBase + r) * KTOT + K0 + ac * 8);
            int gt = tBase + r + kb - 1;
            if ((unsigned)gt < (unsigned)TLEN)
                breg[u] = *(const int4*)(xp + ((size_t)n * TLEN + gt) * FEAT + fi0 + ac * 8);
            else
                breg[u] = make_int4(0, 0, 0, 0);
        }
    };
    auto sstore = [&](int buf) {
        _Float16* A = smem + buf * 5120;
        _Float16* B = smem + 10240 + buf * 5120;
#pragma unroll
        for (int u = 0; u < 2; ++u) {
            int r = ar + 64 * u;
            *(int4*)(A + r * 40 + ac * 8) = areg[u];
            *(int4*)(B + r * 40 + ac * 8) = breg[u];
        }
    };
#endif

    auto compute = [&](int buf) {
        const _Float16* A = smem + buf * 5120;
        const _Float16* B = smem + 10240 + buf * 5120;
        // Load all fragments first so ds_load latency overlaps WMMA issue.
        v16h bfrag[2], afrag[4];
#pragma unroll
        for (int tt = 0; tt < 2; ++tt) {
            // B(32x16): lanes 0-15 hold K0..15 of col N=lane, lanes 16-31 K16..31
            const _Float16* bp = B + (wt * 32 + tt * 16 + m) * 40 + hi * 16;
            v8h b0 = *(const v8h*)bp;
            v8h b1 = *(const v8h*)(bp + 8);
            bfrag[tt] = __builtin_shufflevector(b0, b1, 0,1,2,3,4,5,6,7,8,9,10,11,12,13,14,15);
        }
#pragma unroll
        for (int ff = 0; ff < 4; ++ff) {
            // A(16x32): lanes 0-15 row M, K0..15; lanes 16-31 K16..31
            const _Float16* ap = A + (wf * 64 + ff * 16 + m) * 40 + hi * 16;
            v8h a0 = *(const v8h*)ap;
            v8h a1 = *(const v8h*)(ap + 8);
            afrag[ff] = __builtin_shufflevector(a0, a1, 0,1,2,3,4,5,6,7,8,9,10,11,12,13,14,15);
        }
#pragma unroll
        for (int ff = 0; ff < 4; ++ff)
#pragma unroll
            for (int tt = 0; tt < 2; ++tt) {
#if defined(__HIP_DEVICE_COMPILE__)
                acc[ff][tt] = __builtin_amdgcn_wmma_f32_16x16x32_f16(
                    false, afrag[ff], false, bfrag[tt], (short)0, acc[ff][tt], false, false);
#else
                (void)afrag[ff];
#endif
            }
    };

#if USE_ASYNC_LDS
    issue_tile(0, 0);
    __builtin_amdgcn_s_wait_asynccnt(0);
    __syncthreads();
    for (int ks = 0; ks < 96; ++ks) {
        int cur = ks & 1;
        if (ks + 1 < 96) issue_tile(ks + 1, cur ^ 1);   // async copy overlaps compute
        compute(cur);
        if (ks + 1 < 96) {
            __builtin_amdgcn_s_wait_asynccnt(0);
            __syncthreads();
        }
    }
#else
    gload(0);
    sstore(0);
    __syncthreads();
    for (int ks = 0; ks < 96; ++ks) {
        int cur = ks & 1;
        if (ks + 1 < 96) gload(ks + 1);
        compute(cur);
        if (ks + 1 < 96) {
            sstore(cur ^ 1);
            __syncthreads();
        }
    }
#endif

    // Epilogue: transpose C frags through LDS -> coalesced [t][fo] f16 stores.
    __syncthreads();
#pragma unroll
    for (int ff = 0; ff < 4; ++ff)
#pragma unroll
        for (int tt = 0; tt < 2; ++tt)
#pragma unroll
            for (int r = 0; r < 8; ++r) {
                int M  = hi * 8 + r;          // C layout: VGPR r, lanes 0-15 M=r, 16-31 M=r+8
                int tl = wt * 32 + tt * 16 + m;
                int fl = wf * 64 + ff * 16 + M;
                smem[tl * 132 + fl] = (_Float16)acc[ff][tt][r];
            }
    __syncthreads();

    const float* gvn = gv ? gv + (size_t)n * FEAT : (const float*)0;
#pragma unroll 4
    for (int i = 0; i < 32; ++i) {
        int cid = tid + 256 * i;              // 8192 uint-chunks (2 f16 each)
        int row = cid >> 6;
        int cu  = cid & 63;
        int foc = foBase + cu * 2;
        v2h hv = *(const v2h*)(smem + row * 132 + cu * 2);
        float v0 = (float)hv.x + bias[foc];
        float v1 = (float)hv.y + bias[foc + 1];
        v0 = fmaxf(v0, 0.f);
        v1 = fmaxf(v1, 0.f);
        size_t oidx = ((size_t)n * TLEN + tBase + row) * FEAT + foc;
        if (mode == 1) {
            v2h gfv = *(const v2h*)(gf + oidx);
            float s0 = 1.f / (1.f + __expf(-gvn[foc] * v0));
            float s1 = 1.f / (1.f + __expf(-gvn[foc + 1] * v1));
            v0 = s0 * (float)gfv.x;
            v1 = s1 * (float)gfv.y;
        }
        v2h ov;
        ov.x = (_Float16)v0;
        ov.y = (_Float16)v1;
        *(v2h*)(out + oidx) = ov;
    }
}

// ---------------------------------------------------------------------------
// Head: 1x1 convs -> cas_raw, cas softmax, aas sigmoid, seg_score
// one thread per (n,t)
// ---------------------------------------------------------------------------
__global__ __launch_bounds__(256) void head_kernel(
    const _Float16* __restrict__ hcr, const _Float16* __restrict__ hcf,
    const _Float16* __restrict__ har, const _Float16* __restrict__ haf,
    const float* __restrict__ w2cr, const float* __restrict__ b2cr,
    const float* __restrict__ w2cf, const float* __restrict__ b2cf,
    const float* __restrict__ w2ar, const float* __restrict__ b2ar,
    const float* __restrict__ w2af, const float* __restrict__ b2af,
    float* __restrict__ cas_raw, float* __restrict__ out_aas, float* __restrict__ out_seg)
{
    int id = blockIdx.x * 256 + threadIdx.x;      // NBAT*TLEN
    const _Float16* p1 = hcr + (size_t)id * FEAT;
    const _Float16* p2 = hcf + (size_t)id * FEAT;
    const _Float16* p3 = har + (size_t)id * FEAT;
    const _Float16* p4 = haf + (size_t)id * FEAT;
    float acc[NCLS];
#pragma unroll
    for (int c = 0; c < NCLS; ++c) acc[c] = 0.f;
    float aacc = 0.f;
    for (int fi = 0; fi < FEAT; ++fi) {
        float h1 = (float)p1[fi], h2 = (float)p2[fi];
        float h3 = (float)p3[fi], h4 = (float)p4[fi];
#pragma unroll
        for (int c = 0; c < NCLS; ++c)
            acc[c] += w2cr[c * FEAT + fi] * h1 + w2cf[c * FEAT + fi] * h2;
        aacc += w2ar[fi] * h3 + w2af[fi] * h4;
    }
    float aas   = aacc + b2ar[0] + b2af[0];
    float aas_s = 1.f / (1.f + __expf(-aas));
    out_aas[id] = aas_s;

    float casv[NCLS];
    float mx = -3.4e38f;
#pragma unroll
    for (int c = 0; c < NCLS; ++c) {
        casv[c] = acc[c] + b2cr[c] + b2cf[c];
        cas_raw[(size_t)id * NCLS + c] = casv[c];
        mx = fmaxf(mx, casv[c]);
    }
    float se = 0.f;
#pragma unroll
    for (int c = 0; c < NCLS; ++c) { casv[c] = __expf(casv[c] - mx); se += casv[c]; }
    float inv = 1.f / se;
#pragma unroll
    for (int c = 0; c < NCLS; ++c)
        out_seg[(size_t)id * NCLS + c] = 0.5f * (casv[c] * inv + aas_s);
}

// ---------------------------------------------------------------------------
// Per (n,c) row: stable descending sort (bitonic on (val,idx)), serial
// clustering scan, mask scatter, act/bkg weighted reductions.
// ---------------------------------------------------------------------------
__global__ __launch_bounds__(256) void cluster_kernel(
    const float* __restrict__ seg, const float* __restrict__ cas_raw,
    float* __restrict__ out_mask, float* __restrict__ actArr, float* __restrict__ bkgArr)
{
    __shared__ float sval[TLEN];
    __shared__ int   sidx[TLEN];
    __shared__ float smask[TLEN];
    __shared__ float rbuf[256];

    int row = blockIdx.x;
    int n = row / NCLS, c = row % NCLS;
    int tid = threadIdx.x;

    for (int i = tid; i < TLEN; i += 256) {
        sval[i] = seg[((size_t)n * TLEN + i) * NCLS + c];
        sidx[i] = i;
    }
    __syncthreads();

    // bitonic sort: "ascending" w.r.t. before(a,b) = (a.val>b.val)||(==, a.idx<b.idx)
    for (int k = 2; k <= TLEN; k <<= 1) {
        for (int j = k >> 1; j > 0; j >>= 1) {
            for (int q = tid; q < TLEN / 2; q += 256) {
                int i = ((q & ~(j - 1)) << 1) | (q & (j - 1));
                int l = i | j;
                bool dirUp = ((i & k) == 0);
                float vi = sval[i], vl = sval[l];
                int   ii = sidx[i], il = sidx[l];
                bool lFirst = (vl > vi) || (vl == vi && il < ii);
                if (dirUp == lFirst) {
                    sval[i] = vl; sval[l] = vi;
                    sidx[i] = il; sidx[l] = ii;
                }
            }
            __syncthreads();
        }
    }

    // serial clustering scan
    if (tid == 0) {
        float pos_sum = sval[0], pos_num = 1.f;
        float neg_sum = sval[TLEN - 1], neg_num = 1.f;
        smask[0] = 1.f;
        smask[TLEN - 1] = 0.f;
        for (int i = 0; i < TLEN - 2; ++i) {
            float v = sval[1 + i];
            float w = 1.f / (float)(i + 2);
            w = w * w; w = w * w;                 // (1/(i+2))^4, R=4
            float cf = (fabsf(v - pos_sum / pos_num) <= fabsf(v - neg_sum / neg_num)) ? 1.f : 0.f;
            pos_num += cf * w;
            pos_sum += cf * v * w;
            neg_num += 1.f - cf;
            neg_sum += (1.f - cf) * v;
            smask[1 + i] = cf;
        }
    }
    __syncthreads();

    // scatter mask + partial act/bkg sums
    float an = 0.f, as = 0.f, bn = 0.f, bs = 0.f;
    for (int e = tid; e < TLEN; e += 256) {
        int to = sidx[e];
        float mval = smask[e];
        out_mask[((size_t)n * TLEN + to) * NCLS + c] = mval;
        float rank = 1.f / (float)(e + 1);
        rank = rank * rank; rank = rank * rank;   // (1/(e+1))^4
        float sc = cas_raw[((size_t)n * TLEN + to) * NCLS + c];
        an += rank * mval;
        as += sc * rank * mval;
        bn += 1.f - mval;
        bs += sc * (1.f - mval);
    }

    auto reduceOne = [&](float v) -> float {
        rbuf[tid] = v;
        __syncthreads();
        for (int s = 128; s > 0; s >>= 1) {
            if (tid < s) rbuf[tid] += rbuf[tid + s];
            __syncthreads();
        }
        float r = rbuf[0];
        __syncthreads();
        return r;
    };
    float AN = reduceOne(an);
    float AS = reduceOne(as);
    float BN = reduceOne(bn);
    float BS = reduceOne(bs);
    if (tid == 0) {
        actArr[row] = AS / fmaxf(AN, 1.f);
        bkgArr[row] = BS / fmaxf(BN, 1.f);
    }
}

// ---------------------------------------------------------------------------
// Final per-n softmax over classes for act and bkg
// ---------------------------------------------------------------------------
__global__ void final_softmax_kernel(
    const float* __restrict__ actArr, const float* __restrict__ bkgArr,
    float* __restrict__ out_act, float* __restrict__ out_bkg)
{
    int tid = threadIdx.x;                        // 32 threads
    const float* src = (tid < 16) ? actArr : bkgArr;
    float* dst = (tid < 16) ? out_act : out_bkg;
    int n = tid & 15;
    float v[NCLS];
    float mx = -3.4e38f;
#pragma unroll
    for (int c = 0; c < NCLS; ++c) { v[c] = src[n * NCLS + c]; mx = fmaxf(mx, v[c]); }
    float s = 0.f;
#pragma unroll
    for (int c = 0; c < NCLS; ++c) { v[c] = __expf(v[c] - mx); s += v[c]; }
    float inv = 1.f / s;
#pragma unroll
    for (int c = 0; c < NCLS; ++c) dst[n * NCLS + c] = v[c] * inv;
}

// ---------------------------------------------------------------------------
extern "C" void kernel_launch(void* const* d_in, const int* in_sizes, int n_in,
                              void* d_out, int out_size, void* d_ws, size_t ws_size,
                              hipStream_t stream)
{
    (void)in_sizes; (void)n_in; (void)out_size; (void)ws_size;

    const float* x          = (const float*)d_in[0];
    const float* rgb_gc_w   = (const float*)d_in[1];
    const float* rgb_gc_b   = (const float*)d_in[2];
    const float* rgb_lc_w   = (const float*)d_in[3];
    const float* rgb_lc_b   = (const float*)d_in[4];
    const float* flow_gc_w  = (const float*)d_in[5];
    const float* flow_gc_b  = (const float*)d_in[6];
    const float* flow_lc_w  = (const float*)d_in[7];
    const float* flow_lc_b  = (const float*)d_in[8];
    const float* cas_rgb_w1 = (const float*)d_in[9];
    const float* cas_rgb_b1 = (const float*)d_in[10];
    const float* cas_rgb_w2 = (const float*)d_in[11];
    const float* cas_rgb_b2 = (const float*)d_in[12];
    const float* cas_flow_w1= (const float*)d_in[13];
    const float* cas_flow_b1= (const float*)d_in[14];
    const float* cas_flow_w2= (const float*)d_in[15];
    const float* cas_flow_b2= (const float*)d_in[16];
    const float* aas_rgb_w1 = (const float*)d_in[17];
    const float* aas_rgb_b1 = (const float*)d_in[18];
    const float* aas_rgb_w2 = (const float*)d_in[19];
    const float* aas_rgb_b2 = (const float*)d_in[20];
    const float* aas_flow_w1= (const float*)d_in[21];
    const float* aas_flow_b1= (const float*)d_in[22];
    const float* aas_flow_w2= (const float*)d_in[23];
    const float* aas_flow_b2= (const float*)d_in[24];

    // workspace carve-up
    size_t off = 0;
    auto alloc = [&](size_t bytes) -> void* {
        void* p = (char*)d_ws + off;
        off += (bytes + 255) & ~(size_t)255;
        return p;
    };
    const size_t FEATB = (size_t)NBAT * TLEN * FEAT * sizeof(_Float16);
    _Float16* wpk[6];
    for (int i = 0; i < 6; ++i) wpk[i] = (_Float16*)alloc((size_t)FEAT * KTOT * sizeof(_Float16));
    _Float16* rgb16  = (_Float16*)alloc(FEATB);
    _Float16* flow16 = (_Float16*)alloc(FEATB);
    _Float16* rgbE   = (_Float16*)alloc(FEATB);
    _Float16* flowE  = (_Float16*)alloc(FEATB);
    _Float16* hcr    = (_Float16*)alloc(FEATB);
    _Float16* hcf    = (_Float16*)alloc(FEATB);
    _Float16* har    = (_Float16*)alloc(FEATB);
    _Float16* haf    = (_Float16*)alloc(FEATB);
    float* means   = (float*)alloc((size_t)NBAT * 2048 * sizeof(float));
    float* g_rgb   = (float*)alloc((size_t)NBAT * FEAT * sizeof(float));
    float* g_flow  = (float*)alloc((size_t)NBAT * FEAT * sizeof(float));
    float* cas_raw = (float*)alloc((size_t)NBAT * TLEN * NCLS * sizeof(float));
    float* actArr  = (float*)alloc((size_t)NBAT * NCLS * sizeof(float));
    float* bkgArr  = (float*)alloc((size_t)NBAT * NCLS * sizeof(float));

    // output layout (flat, return order): act(320) bkg(320) aas(16384) seg(327680) mask(327680)
    float* out_act  = (float*)d_out;
    float* out_bkg  = out_act + 320;
    float* out_aas  = out_bkg + 320;
    float* out_seg  = out_aas + NBAT * TLEN;
    float* out_mask = out_seg + (size_t)NBAT * TLEN * NCLS;

    dim3 b256(256);

    // 1) pack activations and weights
    pack_x_kernel<<<(NBAT * TLEN * (FEAT / 4) + 255) / 256, b256, 0, stream>>>(x, rgb16, flow16);
    const float* wsrc[6] = {rgb_lc_w, flow_lc_w, cas_rgb_w1, cas_flow_w1, aas_rgb_w1, aas_flow_w1};
    for (int i = 0; i < 6; ++i)
        pack_w_kernel<<<(FEAT * KTOT + 255) / 256, b256, 0, stream>>>(wsrc[i], wpk[i]);

    // 2) CCA global branch (mean -> GEMV -> relu)
    mean_kernel<<<NBAT, b256, 0, stream>>>(x, means);
    gvec_kernel<<<(NBAT * FEAT + 255) / 256, b256, 0, stream>>>(
        rgb_gc_w, rgb_gc_b, flow_gc_w, flow_gc_b, means, g_rgb, g_flow);

    // 3) WMMA convs
    dim3 cgrid(TLEN / 128, FEAT / 128, NBAT);
    // CCA local branches + gating: rgb_e = sigmoid(g_rgb * relu(conv(flow)+b)) * rgb
    conv3_wmma_kernel<<<cgrid, b256, 0, stream>>>(flow16, wpk[0], rgb_lc_b,  g_rgb,  rgb16,  rgbE, 1);
    conv3_wmma_kernel<<<cgrid, b256, 0, stream>>>(rgb16,  wpk[1], flow_lc_b, g_flow, flow16, flowE, 1);
    // encoder stage-1 convs (relu)
    conv3_wmma_kernel<<<cgrid, b256, 0, stream>>>(rgbE,  wpk[2], cas_rgb_b1,  (const float*)0, (const _Float16*)0, hcr, 0);
    conv3_wmma_kernel<<<cgrid, b256, 0, stream>>>(flowE, wpk[3], cas_flow_b1, (const float*)0, (const _Float16*)0, hcf, 0);
    conv3_wmma_kernel<<<cgrid, b256, 0, stream>>>(rgbE,  wpk[4], aas_rgb_b1,  (const float*)0, (const _Float16*)0, har, 0);
    conv3_wmma_kernel<<<cgrid, b256, 0, stream>>>(flowE, wpk[5], aas_flow_b1, (const float*)0, (const _Float16*)0, haf, 0);

    // 4) head (1x1 convs, softmax, sigmoid, seg_score)
    head_kernel<<<NBAT * TLEN / 256, b256, 0, stream>>>(
        hcr, hcf, har, haf,
        cas_rgb_w2, cas_rgb_b2, cas_flow_w2, cas_flow_b2,
        aas_rgb_w2, aas_rgb_b2, aas_flow_w2, aas_flow_b2,
        cas_raw, out_aas, out_seg);

    // 5) per-(n,c) clustering + scoring
    cluster_kernel<<<NBAT * NCLS, b256, 0, stream>>>(out_seg, cas_raw, out_mask, actArr, bkgArr);

    // 6) final softmax over classes
    final_softmax_kernel<<<1, 32, 0, stream>>>(actArr, bkgArr, out_act, out_bkg);
}
